// BMM_S8T_S8N_S8T_24859270709253
// MI455X (gfx1250) — 8-bit matrix operand order repaired and checked by reading, compiled, not run on hardware
//
#include <hip/hip_runtime.h>
#include <stdint.h>

typedef __attribute__((ext_vector_type(8))) int v8i;

#define BATCH 128
#define MDIM  1024
#define NDIM  1024
#define KDIM  128

__global__ __launch_bounds__(256) void BMM_S8T_S8N_S8T_kernel(
    const int8_t* __restrict__ A,
    const int8_t* __restrict__ Bm,
    int8_t* __restrict__ Y,
    const float* __restrict__ alphaPtr)
{
#pragma clang fp contract(off)

    const float alpha = alphaPtr[0];
    const float MAGIC = 12582912.0f;
    const int   BIAS  = 0x4B400000;
    const int   LO    = BIAS - 128;
    const int   HI    = BIAS + 127;

    const int lane = threadIdx.x & 31;
    const int wave = threadIdx.x >> 5;
    const int half = lane >> 4;
    const int l16  = lane & 15;

    const int bi    = blockIdx.z;
    const int nTile = blockIdx.y * 8 + wave;
    const int m0    = blockIdx.x * 64;

    const int8_t* bRow = Bm + ((size_t)bi * NDIM + (size_t)nTile * 16 + l16) * KDIM;
    v8i bV[2];
#pragma unroll
    for (int j = 0; j < 2; ++j) {
#pragma unroll
        for (int c = 0; c < 4; ++c) {
            const int off = j * 64 + (c >> 1) * 32 + half * 16 + (c & 1) * 8;
            int2 v = *reinterpret_cast<const int2*>(bRow + off);
            bV[j][2 * c]     = v.x;
            bV[j][2 * c + 1] = v.y;
        }
    }

#pragma unroll
    for (int t = 0; t < 4; ++t) {
        const int mBase = m0 + t * 16;
        const int8_t* aRow = A + ((size_t)bi * MDIM + (size_t)mBase + l16) * KDIM;

        v8i aV[2];
#pragma unroll
        for (int j = 0; j < 2; ++j) {
#pragma unroll
            for (int h = 0; h < 2; ++h) {
                const int off = j * 64 + h * 32 + half * 16;
                int4 v = *reinterpret_cast<const int4*>(aRow + off);
                aV[j][4 * h + 0] = v.x;
                aV[j][4 * h + 1] = v.y;
                aV[j][4 * h + 2] = v.z;
                aV[j][4 * h + 3] = v.w;
            }
        }

        v8i acc = {};
        acc = __builtin_amdgcn_wmma_i32_16x16x64_iu8(true, bV[0], true, aV[0], acc, false, false);
        acc = __builtin_amdgcn_wmma_i32_16x16x64_iu8(true, bV[1], true, aV[1], acc, false, false);

        uint32_t q[8];
#pragma unroll
        for (int r = 0; r < 8; ++r) {
            float f = alpha * (float)acc[r];
            f = f + MAGIC;
            int u = __float_as_int(f);
            u = u < LO ? LO : u;
            u = u > HI ? HI : u;
            q[r] = (uint32_t)u;
        }

        uint32_t lo = (q[0] & 0xffu)
                    | ((q[1] & 0xffu) << 8)
                    | ((q[2] & 0xffu) << 16)
                    | (q[3] << 24);
        uint32_t hi = (q[4] & 0xffu)
                    | ((q[5] & 0xffu) << 8)
                    | ((q[6] & 0xffu) << 16)
                    | (q[7] << 24);

        int8_t* yPtr = Y + ((size_t)bi * MDIM + (size_t)mBase + l16) * NDIM
                         + (size_t)nTile * 16 + half * 8;
        uint2 packed; packed.x = lo; packed.y = hi;
        *reinterpret_cast<uint2*>(yPtr) = packed;
    }
}

extern "C" void kernel_launch(void* const* d_in, const int* in_sizes, int n_in,
                              void* d_out, int out_size, void* d_ws, size_t ws_size,
                              hipStream_t stream) {
    const int8_t* a     = (const int8_t*)d_in[0];
    const int8_t* b     = (const int8_t*)d_in[1];
    const float*  alpha = (const float*)d_in[2];
    int8_t*       y     = (int8_t*)d_out;

    (void)in_sizes; (void)n_in; (void)out_size; (void)d_ws; (void)ws_size;

    dim3 grid(MDIM / 64, (NDIM / 16) / 8, BATCH);
    dim3 block(256);
    BMM_S8T_S8N_S8T_kernel<<<grid, block, 0, stream>>>(a, b, y, alpha);
}
